// CausalSelfAttention_76553497084442
// MI455X (gfx1250) — matrix operand order repaired and checked by reading, not compiled, not run on hardware
//
#include <hip/hip_runtime.h>
#include <stdint.h>

typedef __bf16 bf16;
typedef __attribute__((ext_vector_type(16))) __bf16 v16bf;
typedef __attribute__((ext_vector_type(8)))  float  v8f;
typedef __attribute__((ext_vector_type(4)))  int    v4i;

#define WMMA_BF16(a, b, c) \
  __builtin_amdgcn_wmma_f32_16x16x32_bf16(false, (a), false, (b), (short)0, (c), false, false)

// ---------------------------------------------------------------------------
// CDNA5 async global->LDS DMA (GLOBAL_LOAD_ASYNC_TO_LDS_B128), guarded so the
// kernel still builds (with the VGPR round-trip path) if the builtin is absent.
// Builtin prototype (from hipcc diagnostic): (int4 AS1*, int4 AS3*, imm, imm).
// ---------------------------------------------------------------------------
#if defined(__has_builtin)
#if __has_builtin(__builtin_amdgcn_global_load_async_to_lds_b128)
#define HAVE_ASYNC_LDS 1
#endif
#endif
#ifndef HAVE_ASYNC_LDS
#define HAVE_ASYNC_LDS 0
#endif

#if HAVE_ASYNC_LDS
typedef __attribute__((address_space(1))) v4i gv4i;  // global int4
typedef __attribute__((address_space(3))) v4i lv4i;  // LDS int4
__device__ inline void async_copy_b128(const void* g, void* l) {
  __builtin_amdgcn_global_load_async_to_lds_b128((gv4i*)g, (lv4i*)l, 0, 0);
}
__device__ inline void wait_async0() {
#if __has_builtin(__builtin_amdgcn_s_wait_asynccnt)
  __builtin_amdgcn_s_wait_asynccnt(0);
#else
  asm volatile("s_wait_asynccnt 0x0" ::: "memory");
#endif
}
#endif

// ---------------------------------------------------------------------------
// Fragment loaders (LDS -> VGPR) following CDNA5 ISA 7.12.2 layouts.
// A 16x32 bf16: lane m=lane&15; lanes>=16 get K+8; VGPR p holds K pair
//   {(p>=4?16:0)+2*(p&3), +1}.
// B 32x16 bf16: lane n=lane&15; same K order as A: lanes>=16 get K+8,
//   elements 0-7 hold K 0..7, elements 8-15 hold K 16..23 (two b128 LDS loads).
// ---------------------------------------------------------------------------
__device__ inline v16bf ld_frag_a(const bf16* p, int ld, int lane) {
  union { v16bf v; uint32_t u[8]; } f;
  const bf16* row = p + (lane & 15) * ld + ((lane >> 4) << 3);
#pragma unroll
  for (int q = 0; q < 8; ++q) {
    const int k = ((q & 4) << 2) | ((q & 3) << 1);
    f.u[q] = *(const uint32_t*)(row + k);
  }
  return f.v;
}

__device__ inline v16bf ld_frag_b(const bf16* p, int ld, int lane) {
  union { v16bf v; uint4 q[2]; } f;
  const bf16* row = p + (lane & 15) * ld + ((lane >> 4) << 3);
  f.q[0] = *(const uint4*)(row);
  f.q[1] = *(const uint4*)(row + 16);
  return f.v;
}

// ---------------------------------------------------------------------------
// fp32 -> bf16 convert
// ---------------------------------------------------------------------------
__global__ __launch_bounds__(256) void cvt_f32_to_bf16(const float* __restrict__ in,
                                                       bf16* __restrict__ out, int n) {
  const int i = (blockIdx.x * blockDim.x + threadIdx.x) * 4;
  if (i + 3 < n) {
    const float4 f = *(const float4*)(in + i);
    struct alignas(8) B4 { bf16 a, b, c, d; };
    B4 v{(bf16)f.x, (bf16)f.y, (bf16)f.z, (bf16)f.w};
    *(B4*)(out + i) = v;
  }
}

// ---------------------------------------------------------------------------
// C = A(MxK) @ B(KxN) + bias, bf16 inputs, f32 accumulate, OutT output.
// Block tile 128x128, 8 waves, wave tile 32x64, K step 32.
// Double-buffered LDS: tile k+1 streams in (A via async DMA, B via VGPR
// transpose) while tile k is consumed by WMMA. One barrier per k-step.
// ---------------------------------------------------------------------------
template <typename OutT>
__global__ __launch_bounds__(256) void gemm_bf16_wmma(
    const bf16* __restrict__ A, const bf16* __restrict__ B,
    const float* __restrict__ bias, OutT* __restrict__ C,
    int M, int N, int K) {
  __shared__ bf16 As[2][128][40];  // row-major tile, stride 40 elems (80B rows)
  __shared__ bf16 Bs[2][128][40];  // transposed tile: Bs[b][n][k]

  const int tid  = threadIdx.x;
  const int lane = tid & 31;
  const int wid  = tid >> 5;
  const int wm   = (wid & 3) * 32;
  const int wn   = (wid >> 2) * 64;
  const int bm   = blockIdx.y * 128;
  const int bn   = blockIdx.x * 128;

  v8f acc[2][4];
#pragma unroll
  for (int i = 0; i < 2; ++i)
#pragma unroll
    for (int j = 0; j < 4; ++j) acc[i][j] = (v8f){0, 0, 0, 0, 0, 0, 0, 0};

  const int arow = tid >> 1, acb = (tid & 1) * 16;  // A: 128 rows x 32 cols
  const int brow = tid >> 3, bcb = (tid & 7) * 16;  // B: 32 rows x 128 cols
  const bf16* gA = A + (size_t)(bm + arow) * K + acb;

  // ---- prologue: stage tile 0 into buffer 0 ----
  {
#if HAVE_ASYNC_LDS
    async_copy_b128(gA,     &As[0][arow][acb]);
    async_copy_b128(gA + 8, &As[0][arow][acb + 8]);
#else
    const uint4* ga = (const uint4*)gA;
    *(uint4*)&As[0][arow][acb]     = ga[0];
    *(uint4*)&As[0][arow][acb + 8] = ga[1];
#endif
    const uint4* gb = (const uint4*)(B + (size_t)brow * N + bn + bcb);
    union { uint4 q; bf16 h[8]; } u0, u1;
    u0.q = gb[0]; u1.q = gb[1];
#pragma unroll
    for (int e = 0; e < 8; ++e) Bs[0][bcb + e][brow]     = u0.h[e];
#pragma unroll
    for (int e = 0; e < 8; ++e) Bs[0][bcb + 8 + e][brow] = u1.h[e];
  }
#if HAVE_ASYNC_LDS
  wait_async0();
#endif
  __syncthreads();

  int cur = 0;
  for (int k0 = 0; k0 < K; k0 += 32) {
    const int nxt = cur ^ 1;
    const bool more = (k0 + 32) < K;
    uint4 a0 = {}, a1 = {}, b0 = {}, b1 = {};
    if (more) {  // issue next tile's traffic before computing
      const bf16* ga = gA + (k0 + 32);
#if HAVE_ASYNC_LDS
      async_copy_b128(ga,     &As[nxt][arow][acb]);
      async_copy_b128(ga + 8, &As[nxt][arow][acb + 8]);
#else
      a0 = ((const uint4*)ga)[0];
      a1 = ((const uint4*)ga)[1];
#endif
      const uint4* gb = (const uint4*)(B + (size_t)(k0 + 32 + brow) * N + bn + bcb);
      b0 = gb[0]; b1 = gb[1];
    }

    // ---- compute on current buffer ----
    const v16bf af0 = ld_frag_a(&As[cur][wm][0],      40, lane);
    const v16bf af1 = ld_frag_a(&As[cur][wm + 16][0], 40, lane);
#pragma unroll
    for (int j = 0; j < 4; ++j) {
      const v16bf bf = ld_frag_b(&Bs[cur][wn + j * 16][0], 40, lane);
      acc[0][j] = WMMA_BF16(af0, bf, acc[0][j]);
      acc[1][j] = WMMA_BF16(af1, bf, acc[1][j]);
    }

    if (more) {  // drain next tile into LDS (safe: nobody reads nxt yet)
#if !HAVE_ASYNC_LDS
      *(uint4*)&As[nxt][arow][acb]     = a0;
      *(uint4*)&As[nxt][arow][acb + 8] = a1;
#endif
      union { uint4 q; bf16 h[8]; } u0, u1;
      u0.q = b0; u1.q = b1;
#pragma unroll
      for (int e = 0; e < 8; ++e) Bs[nxt][bcb + e][brow]     = u0.h[e];
#pragma unroll
      for (int e = 0; e < 8; ++e) Bs[nxt][bcb + 8 + e][brow] = u1.h[e];
    }
#if HAVE_ASYNC_LDS
    wait_async0();
#endif
    __syncthreads();
    cur = nxt;
  }

  // ---- epilogue ----
  const int half8 = (lane >> 4) << 3;
  const int ln    = lane & 15;
#pragma unroll
  for (int j = 0; j < 4; ++j) {
    const int n = bn + wn + j * 16 + ln;
    const float bv = bias ? bias[n] : 0.0f;
#pragma unroll
    for (int i = 0; i < 2; ++i) {
#pragma unroll
      for (int r = 0; r < 8; ++r) {
        const int m = bm + wm + i * 16 + half8 + r;
        C[(size_t)m * N + n] = (OutT)(acc[i][j][r] + bv);
      }
    }
  }
}

// ---------------------------------------------------------------------------
// Flash attention forward. qkv: (B*T, 3C) bf16, layout [q|k|v] per row.
// One block = (b, h, 64-row q tile). 4 waves x 16 q rows. hd = 64.
// Q/K tiles stream global->LDS via async DMA (straight copy); V goes through
// VGPRs because it is transposed into LDS.
// ---------------------------------------------------------------------------
__global__ __launch_bounds__(128) void attn_fwd(const bf16* __restrict__ qkv,
                                                bf16* __restrict__ y) {
  const int T = 2048, C3 = 3072, Cc = 1024, HD = 64;
  __shared__ bf16 Qs[64][72];  // stride 72 elems = 144B (16B aligned)
  __shared__ bf16 Ks[64][72];  // K tile row-major: Ks[kpos][hd]
  __shared__ bf16 Vt[64][72];  // V tile transposed: Vt[hd][kpos]
  __shared__ bf16 Ps[64][72];  // probabilities round-trip (C->A layout)

  const int tid   = threadIdx.x;
  const int lane  = tid & 31;
  const int w     = tid >> 5;    // wave 0..3
  const int qb    = blockIdx.x;  // q tile 0..31
  const int h     = blockIdx.y;
  const int b     = blockIdx.z;
  const int half8 = (lane >> 4) << 3;
  const int ln    = lane & 15;

  const size_t baserow = (size_t)b * T;
  const int qcol = h * HD;
  const int kcol = Cc + h * HD;
  const int vcol = 2 * Cc + h * HD;
  const int r2 = tid >> 1, cb2 = (tid & 1) * 32;  // cooperative 64x64 tiling

  {  // cooperative Q tile load (64x64)
    const bf16* g = qkv + (baserow + qb * 64 + r2) * C3 + qcol + cb2;
#pragma unroll
    for (int e = 0; e < 4; ++e) {
#if HAVE_ASYNC_LDS
      async_copy_b128(g + 8 * e, &Qs[r2][cb2 + 8 * e]);
#else
      *(uint4*)&Qs[r2][cb2 + 8 * e] = ((const uint4*)g)[e];
#endif
    }
  }
#if HAVE_ASYNC_LDS
  wait_async0();
#endif
  __syncthreads();
  const v16bf qf0 = ld_frag_a(&Qs[w * 16][0],  72, lane);
  const v16bf qf1 = ld_frag_a(&Qs[w * 16][32], 72, lane);

  float mrow[8], lrow[8];
  v8f o[4];
#pragma unroll
  for (int r = 0; r < 8; ++r) { mrow[r] = -1e30f; lrow[r] = 0.f; }
#pragma unroll
  for (int j = 0; j < 4; ++j) o[j] = (v8f){0, 0, 0, 0, 0, 0, 0, 0};

  const float scale = 0.125f;  // 1/sqrt(64)

  for (int kb = 0; kb <= qb; ++kb) {
    __syncthreads();  // all waves done with previous K/V tiles
    {
      const bf16* gk = qkv + (baserow + kb * 64 + r2) * C3 + kcol + cb2;
#pragma unroll
      for (int e = 0; e < 4; ++e) {
#if HAVE_ASYNC_LDS
        async_copy_b128(gk + 8 * e, &Ks[r2][cb2 + 8 * e]);
#else
        *(uint4*)&Ks[r2][cb2 + 8 * e] = ((const uint4*)gk)[e];
#endif
      }
      const uint4* gv = (const uint4*)(qkv + (baserow + kb * 64 + r2) * C3 + vcol + cb2);
#pragma unroll
      for (int e = 0; e < 4; ++e) {
        union { uint4 q; bf16 hh[8]; } u; u.q = gv[e];
#pragma unroll
        for (int x = 0; x < 8; ++x) Vt[cb2 + 8 * e + x][r2] = u.hh[x];
      }
    }
#if HAVE_ASYNC_LDS
    wait_async0();
#endif
    __syncthreads();

    // S = Q @ K^T : B-fragment of K^T = row-contiguous read of Ks
    v8f s[4];
#pragma unroll
    for (int j = 0; j < 4; ++j) {
      const v16bf kf0 = ld_frag_b(&Ks[j * 16][0],  72, lane);
      const v16bf kf1 = ld_frag_b(&Ks[j * 16][32], 72, lane);
      v8f z = (v8f){0, 0, 0, 0, 0, 0, 0, 0};
      z = WMMA_BF16(qf0, kf0, z);
      z = WMMA_BF16(qf1, kf1, z);
      s[j] = z;
    }

    // scale + causal mask
    const int qrow0 = qb * 64 + w * 16 + half8;
#pragma unroll
    for (int j = 0; j < 4; ++j) {
      const int kpos = kb * 64 + j * 16 + ln;
#pragma unroll
      for (int r = 0; r < 8; ++r) {
        float v = s[j][r] * scale;
        if (kpos > qrow0 + r) v = -1e30f;
        s[j][r] = v;
      }
    }

    // online softmax: per-row reductions stay inside each 16-lane half
#pragma unroll
    for (int r = 0; r < 8; ++r) {
      float mx = fmaxf(fmaxf(s[0][r], s[1][r]), fmaxf(s[2][r], s[3][r]));
#pragma unroll
      for (int off = 8; off >= 1; off >>= 1) mx = fmaxf(mx, __shfl_xor(mx, off, 32));
      const float mnew = fmaxf(mrow[r], mx);
      const float corr = __expf(mrow[r] - mnew);
      mrow[r] = mnew;
      float rs = 0.f;
#pragma unroll
      for (int j = 0; j < 4; ++j) {
        const float p = __expf(s[j][r] - mnew);
        s[j][r] = p;
        rs += p;
      }
#pragma unroll
      for (int off = 8; off >= 1; off >>= 1) rs += __shfl_xor(rs, off, 32);
      lrow[r] = lrow[r] * corr + rs;
#pragma unroll
      for (int j = 0; j < 4; ++j) o[j][r] *= corr;
    }

    // C-layout f32 -> A-layout bf16 via LDS (wave-local rows; HW waits DScnt)
#pragma unroll
    for (int j = 0; j < 4; ++j)
#pragma unroll
      for (int r = 0; r < 8; ++r)
        Ps[w * 16 + half8 + r][j * 16 + ln] = (bf16)s[j][r];

    const v16bf pf0 = ld_frag_a(&Ps[w * 16][0],  72, lane);
    const v16bf pf1 = ld_frag_a(&Ps[w * 16][32], 72, lane);
#pragma unroll
    for (int j = 0; j < 4; ++j) {
      const v16bf vf0 = ld_frag_b(&Vt[j * 16][0],  72, lane);
      const v16bf vf1 = ld_frag_b(&Vt[j * 16][32], 72, lane);
      o[j] = WMMA_BF16(pf0, vf0, o[j]);
      o[j] = WMMA_BF16(pf1, vf1, o[j]);
    }
  }

  // y = O / l  (bf16, consumed by projection GEMM)
#pragma unroll
  for (int r = 0; r < 8; ++r) {
    const float inv = 1.0f / lrow[r];
    const size_t t = baserow + qb * 64 + w * 16 + half8 + r;
#pragma unroll
    for (int j = 0; j < 4; ++j)
      y[t * Cc + h * HD + j * 16 + ln] = (bf16)(o[j][r] * inv);
  }
}

// ---------------------------------------------------------------------------
extern "C" void kernel_launch(void* const* d_in, const int* in_sizes, int n_in,
                              void* d_out, int out_size, void* d_ws, size_t ws_size,
                              hipStream_t stream) {
  const float* x      = (const float*)d_in[0];
  const float* W_attn = (const float*)d_in[1];
  const float* b_attn = (const float*)d_in[2];
  const float* W_proj = (const float*)d_in[3];
  const float* b_proj = (const float*)d_in[4];
  float* out = (float*)d_out;

  const int Bb = 4, T = 2048, Cc = 1024, C3 = 3072;
  const int M = Bb * T;  // 8192

  char* ws = (char*)d_ws;
  size_t off = 0;
  auto alloc = [&](size_t bytes) -> void* {
    void* p = ws + off;
    off = (off + bytes + 255) & ~(size_t)255;
    return p;
  };
  bf16* x_bf   = (bf16*)alloc((size_t)M * Cc * 2);   // 16 MB
  bf16* wa_bf  = (bf16*)alloc((size_t)Cc * C3 * 2);  //  6 MB
  bf16* wp_bf  = (bf16*)alloc((size_t)Cc * Cc * 2);  //  2 MB
  bf16* qkv_bf = (bf16*)alloc((size_t)M * C3 * 2);   // 48 MB
  bf16* y_bf   = (bf16*)alloc((size_t)M * Cc * 2);   // 16 MB
  (void)ws_size; (void)in_sizes; (void)n_in; (void)out_size;

  // fp32 -> bf16
  cvt_f32_to_bf16<<<dim3(M * Cc / 1024), dim3(256), 0, stream>>>(x, x_bf, M * Cc);
  cvt_f32_to_bf16<<<dim3(Cc * C3 / 1024), dim3(256), 0, stream>>>(W_attn, wa_bf, Cc * C3);
  cvt_f32_to_bf16<<<dim3(Cc * Cc / 1024), dim3(256), 0, stream>>>(W_proj, wp_bf, Cc * Cc);

  // qkv = x @ W_attn + b_attn   (8192 x 3072)
  gemm_bf16_wmma<bf16><<<dim3(C3 / 128, M / 128), dim3(256), 0, stream>>>(
      x_bf, wa_bf, b_attn, qkv_bf, M, C3, Cc);

  // flash attention
  attn_fwd<<<dim3(T / 64, 16, Bb), dim3(128), 0, stream>>>(qkv_bf, y_bf);

  // out = y @ W_proj + b_proj   (8192 x 1024, fp32)
  gemm_bf16_wmma<float><<<dim3(Cc / 128, M / 128), dim3(256), 0, stream>>>(
      y_bf, wp_bf, b_proj, out, M, Cc, Cc);
}
